// HeteroGATSingleHead_11828339933784
// MI455X (gfx1250) — matrix operand order repaired and checked by reading, not compiled, not run on hardware
//
#include <hip/hip_runtime.h>
#include <hip/hip_bf16.h>
#include <stdint.h>

#define NN 50000   // nodes
#define NE 800000  // edges
#define IFT 256    // in feats
#define OF 64      // out feats
#define NH 4       // heads
#define NC 4       // channels
#define HO 256     // NH*OF

typedef __attribute__((ext_vector_type(16))) __bf16 v16bf;
typedef __attribute__((ext_vector_type(8)))  float  v8f;
typedef __attribute__((__vector_size__(16))) int    v4i;   // matches builtin param

union BfVec { uint4 q[2]; v16bf v; };

#ifndef __has_builtin
#define __has_builtin(x) 0
#endif
#if __has_builtin(__builtin_amdgcn_global_load_async_to_lds_b128)
#define USE_ASYNC_LDS 1
#endif

#define ASG __attribute__((address_space(1)))
#define ASL __attribute__((address_space(3)))

#if USE_ASYNC_LDS
__device__ __forceinline__ void waitAsync0() {
#if __has_builtin(__builtin_amdgcn_s_wait_asynccnt)
  __builtin_amdgcn_s_wait_asynccnt(0);
#else
  asm volatile("s_wait_asynccnt 0x0" ::: "memory");
#endif
}
#endif

__device__ __forceinline__ unsigned short f2bf(float f) {
  unsigned u = __float_as_uint(f);
  unsigned r = 0x7FFFu + ((u >> 16) & 1u);   // round-to-nearest-even
  return (unsigned short)((u + r) >> 16);
}
// order-preserving float<->uint map for atomicMax on signed floats
__device__ __forceinline__ unsigned encOrd(float f) {
  unsigned u = __float_as_uint(f);
  return (u & 0x80000000u) ? ~u : (u | 0x80000000u);
}
__device__ __forceinline__ float decOrd(unsigned u) {
  return __uint_as_float((u & 0x80000000u) ? (u & 0x7FFFFFFFu) : ~u);
}

__global__ void k_fill_u32(unsigned* p, unsigned v, int n) {
  int i = blockIdx.x * blockDim.x + threadIdx.x;
  if (i < n) p[i] = v;
}

__global__ void k_cvt_bf16(const float* __restrict__ src,
                           unsigned short* __restrict__ dst, int n4) {
  int i = blockIdx.x * blockDim.x + threadIdx.x;
  if (i >= n4) return;
  float4 f = ((const float4*)src)[i];
  ushort4 b;
  b.x = f2bf(f.x); b.y = f2bf(f.y); b.z = f2bf(f.z); b.w = f2bf(f.w);
  ((ushort4*)dst)[i] = b;
}

// ---------------- WMMA GEMM: Wh[c] = feature @ Wl[c]^T ----------------
// 256 thr = 8 waves, 64x64 C tile, K-step 32 (bf16), double-buffered LDS,
// async global->LDS staging (ASYNCcnt) overlapped with WMMA when available.
#define LDT 40  // padded LDS stride (bf16 elems) -> conflict-free b128 reads
__global__ void __launch_bounds__(256)
k_gemm(const unsigned short* __restrict__ featB,
       const unsigned short* __restrict__ WB, float* __restrict__ Wh) {
  const int c  = blockIdx.z;
  const int r0 = blockIdx.x * 64;
  const int c0 = blockIdx.y * 64;
  __shared__ __align__(16) unsigned short lA[2][64 * LDT];
  __shared__ __align__(16) unsigned short lB[2][64 * LDT];
  const int t = threadIdx.x;
  const int wave = t >> 5, lane = t & 31;
  const int wr = wave & 3, wc = wave >> 2;          // wave -> 16x32 C tile
  const int lhalf = lane >> 4, lsub = lane & 15;

  v8f acc0 = {}; v8f acc1 = {};
  const unsigned short* Wc = WB + (size_t)c * HO * IFT;

  const int lrow = t >> 2;            // 64 rows, 4 thr/row
  const int lcg  = (t & 3) * 8;       // 8 bf16 (one b128) per thread
  // clamp instead of guard: duplicate rows only feed C rows >= NN (never stored)
  int gr = r0 + lrow; if (gr > NN - 1) gr = NN - 1;
  const unsigned short* gA = featB + (size_t)gr * IFT + lcg;
  const unsigned short* gB = Wc + (size_t)(c0 + lrow) * IFT + lcg;

#if USE_ASYNC_LDS
#define STAGE(buf, k0)                                                         \
  do {                                                                         \
    __builtin_amdgcn_global_load_async_to_lds_b128(                            \
        (ASG v4i*)(gA + (k0)), (ASL v4i*)(&lA[buf][lrow * LDT + lcg]), 0, 0);  \
    __builtin_amdgcn_global_load_async_to_lds_b128(                            \
        (ASG v4i*)(gB + (k0)), (ASL v4i*)(&lB[buf][lrow * LDT + lcg]), 0, 0);  \
  } while (0)
#else
#define STAGE(buf, k0)                                                         \
  do {                                                                         \
    uint4 av = *(const uint4*)(gA + (k0));                                     \
    uint4 bv = *(const uint4*)(gB + (k0));                                     \
    *(uint4*)(&lA[buf][lrow * LDT + lcg]) = av;                                \
    *(uint4*)(&lB[buf][lrow * LDT + lcg]) = bv;                                \
  } while (0)
#endif

  STAGE(0, 0);
#pragma unroll
  for (int kk = 0; kk < 8; ++kk) {
    const int buf = kk & 1;
#if USE_ASYNC_LDS
    waitAsync0();                 // this wave's async tiles have landed
#endif
    __syncthreads();              // everyone's tiles visible; prev reads done
    if (kk < 7) STAGE(buf ^ 1, (kk + 1) * 32);

    // A operand: row=lsub, K contiguous 16 at kb=lhalf*16
    BfVec a, b0, b1;
    {
      const unsigned short* ap = &lA[buf][(wr * 16 + lsub) * LDT + lhalf * 16];
      a.q[0] = *(const uint4*)(ap);
      a.q[1] = *(const uint4*)(ap + 8);
    }
    // B operand: col=lsub, K contiguous 16 at kb=lhalf*16
    {
      const unsigned short* bp = &lB[buf][(wc * 32 + lsub) * LDT + lhalf * 16];
      b0.q[0] = *(const uint4*)(bp);
      b0.q[1] = *(const uint4*)(bp + 8);
      const unsigned short* bq = bp + 16 * LDT;
      b1.q[0] = *(const uint4*)(bq);
      b1.q[1] = *(const uint4*)(bq + 8);
    }
    acc0 = __builtin_amdgcn_wmma_f32_16x16x32_bf16(false, a.v, false, b0.v,
                                                   (short)0, acc0, false, false);
    acc1 = __builtin_amdgcn_wmma_f32_16x16x32_bf16(false, a.v, false, b1.v,
                                                   (short)0, acc1, false, false);
  }
#undef STAGE
  // C layout: vgpr i -> (M=i + 8*lhalf, N=lsub)
  float* out = Wh + (size_t)c * NN * HO;
  const int n  = c0 + wc * 32 + lsub;
  const int mb = r0 + wr * 16 + lhalf * 8;
#pragma unroll
  for (int i = 0; i < 8; ++i) {
    int m = mb + i;
    if (m < NN) {
      out[(size_t)m * HO + n]      = acc0[i];
      out[(size_t)m * HO + n + 16] = acc1[i];
    }
  }
}

// ---------------- el/er = <Wh, a_src/a_dst> per head (wave per node) ----------------
__global__ void __launch_bounds__(256)
k_attn(const float* __restrict__ Wh, const float* __restrict__ a_src,
       const float* __restrict__ a_dst, float* __restrict__ el,
       float* __restrict__ er) {
  const int c = blockIdx.y;
  const int wave = threadIdx.x >> 5, lane = threadIdx.x & 31;
  const int n = blockIdx.x * 8 + wave;
  if (n >= NN) return;
  const float* w  = Wh + ((size_t)c * NN + n) * HO + lane * 8;
  const float* as = a_src + (size_t)c * HO + lane * 8;
  const float* ad = a_dst + (size_t)c * HO + lane * 8;
  float pl = 0.f, pr = 0.f;
#pragma unroll
  for (int j = 0; j < 8; ++j) {
    float wv = w[j];
    pl += wv * as[j];
    pr += wv * ad[j];
  }
#pragma unroll
  for (int m = 1; m < 8; m <<= 1) {    // 8-lane segments = one head each
    pl += __shfl_xor(pl, m, 8);
    pr += __shfl_xor(pr, m, 8);
  }
  if ((lane & 7) == 0) {
    int h = lane >> 3;
    el[((size_t)c * NN + n) * NH + h] = pl;
    er[((size_t)c * NN + n) * NH + h] = pr;
  }
}

// ---------------- edge pass 1: e = leaky(el[s]+er[d]); segment max ----------------
__global__ void k_edge_max(const int* __restrict__ src, const int* __restrict__ dst,
                           const float* __restrict__ el, const float* __restrict__ er,
                           float* __restrict__ eBuf, unsigned* __restrict__ mEnc) {
  const int c = blockIdx.y;
  const int e = blockIdx.x * blockDim.x + threadIdx.x;
  if (e >= NE) return;
  const int s = src[(size_t)c * NE + e];
  const int d = dst[(size_t)c * NE + e];
  float4 l = *(const float4*)(el + ((size_t)c * NN + s) * NH);
  float4 r = *(const float4*)(er + ((size_t)c * NN + d) * NH);
  float4 ev;
  ev.x = l.x + r.x; ev.y = l.y + r.y; ev.z = l.z + r.z; ev.w = l.w + r.w;
  ev.x = ev.x > 0.f ? ev.x : 0.2f * ev.x;
  ev.y = ev.y > 0.f ? ev.y : 0.2f * ev.y;
  ev.z = ev.z > 0.f ? ev.z : 0.2f * ev.z;
  ev.w = ev.w > 0.f ? ev.w : 0.2f * ev.w;
  *(float4*)(eBuf + ((size_t)c * NE + e) * NH) = ev;
  unsigned* mp = mEnc + ((size_t)c * NN + d) * NH;
  atomicMax(mp + 0, encOrd(ev.x));
  atomicMax(mp + 1, encOrd(ev.y));
  atomicMax(mp + 2, encOrd(ev.z));
  atomicMax(mp + 3, encOrd(ev.w));
}

__global__ void k_decode_m(unsigned* m, int n) {
  int i = blockIdx.x * blockDim.x + threadIdx.x;
  if (i >= n) return;
  float f = decOrd(m[i]);
  unsigned b = __float_as_uint(f);
  if ((b & 0x7F800000u) == 0x7F800000u) f = 0.f;   // !isfinite -> 0
  ((float*)m)[i] = f;
}

// ---------------- edge pass 2: ex = exp(e - m[d]); segment sum ----------------
__global__ void k_edge_exp(const int* __restrict__ dst, float* __restrict__ eBuf,
                           const float* __restrict__ m, float* __restrict__ den) {
  const int c = blockIdx.y;
  const int e = blockIdx.x * blockDim.x + threadIdx.x;
  if (e >= NE) return;
  const int d = dst[(size_t)c * NE + e];
  float4 ev = *(float4*)(eBuf + ((size_t)c * NE + e) * NH);
  float4 mv = *(const float4*)(m + ((size_t)c * NN + d) * NH);
  float4 x;
  x.x = expf(ev.x - mv.x); x.y = expf(ev.y - mv.y);
  x.z = expf(ev.z - mv.z); x.w = expf(ev.w - mv.w);
  *(float4*)(eBuf + ((size_t)c * NE + e) * NH) = x;
  float* dp = den + ((size_t)c * NN + d) * NH;
  atomicAdd(dp + 0, x.x);
  atomicAdd(dp + 1, x.y);
  atomicAdd(dp + 2, x.z);
  atomicAdd(dp + 3, x.w);
}

// ---------------- edge pass 3: h[d] += a * Wh[s]  (wave per edge, 1KB coalesced) ----------------
__global__ void __launch_bounds__(256)
k_edge_scatter(const int* __restrict__ src, const int* __restrict__ dst,
               const float* __restrict__ eBuf, const float* __restrict__ den,
               const float* __restrict__ Wh, float* __restrict__ Hacc) {
  const int c = blockIdx.y;
  const int wave = threadIdx.x >> 5, lane = threadIdx.x & 31;
  const int e = blockIdx.x * 8 + wave;
  if (e >= NE) return;
  const int s = src[(size_t)c * NE + e];
  const int d = dst[(size_t)c * NE + e];
  const int h = lane >> 3;   // 8 lanes per head (64 floats)
  float a = eBuf[((size_t)c * NE + e) * NH + h] /
            den[((size_t)c * NN + d) * NH + h];
  const float4* wp = (const float4*)(Wh + ((size_t)c * NN + s) * HO + lane * 8);
  float4 w0 = wp[0], w1 = wp[1];
  float* hp = Hacc + ((size_t)c * NN + d) * HO + lane * 8;
  atomicAdd(hp + 0, a * w0.x);
  atomicAdd(hp + 1, a * w0.y);
  atomicAdd(hp + 2, a * w0.z);
  atomicAdd(hp + 3, a * w0.w);
  atomicAdd(hp + 4, a * w1.x);
  atomicAdd(hp + 5, a * w1.y);
  atomicAdd(hp + 6, a * w1.z);
  atomicAdd(hp + 7, a * w1.w);
}

// ---------------- h = elu(segsum + Wh + bias) ----------------
__global__ void k_elu_bias(float* __restrict__ Hacc, const float* __restrict__ Wh,
                           const float* __restrict__ bias, int n4) {
  int i = blockIdx.x * blockDim.x + threadIdx.x;
  if (i >= n4) return;
  const int perC = NN * (HO / 4);
  int c = i / perC;
  int q = i % (HO / 4);
  float4 h = ((float4*)Hacc)[i];
  float4 w = ((const float4*)Wh)[i];
  float4 b = ((const float4*)bias)[c * (HO / 4) + q];
  float4 r;
  float x;
  x = h.x + w.x + b.x; r.x = x > 0.f ? x : expm1f(x);
  x = h.y + w.y + b.y; r.y = x > 0.f ? x : expm1f(x);
  x = h.z + w.z + b.z; r.z = x > 0.f ? x : expm1f(x);
  x = h.w + w.w + b.w; r.w = x > 0.f ? x : expm1f(x);
  ((float4*)Hacc)[i] = r;
}

__device__ __forceinline__ float dot8(float4 a0, float4 a1, const float* w) {
  return a0.x * w[0] + a0.y * w[1] + a0.z * w[2] + a0.w * w[3] +
         a1.x * w[4] + a1.y * w[5] + a1.z * w[6] + a1.w * w[7];
}

// ---------------- semantic aggregation + type select (wave per node) ----------------
__global__ void __launch_bounds__(256)
k_aggregate(const float* __restrict__ Hacc, const float* __restrict__ Dw,
            const float* __restrict__ Db, const float* __restrict__ Ww,
            const float* __restrict__ Wb, const unsigned char* __restrict__ mask,
            float* __restrict__ out) {
  const int wave = threadIdx.x >> 5, lane = threadIdx.x & 31;
  const int n = blockIdx.x * 8 + wave;
  if (n >= NN) return;
  const size_t stride = (size_t)NN * HO;
  const float* base = Hacc + (size_t)n * HO + lane * 8;
  float4 dd0 = *(const float4*)(base + 0 * stride);
  float4 dd1 = *(const float4*)(base + 0 * stride + 4);
  float4 wd0 = *(const float4*)(base + 1 * stride);
  float4 wd1 = *(const float4*)(base + 1 * stride + 4);
  float4 dw0 = *(const float4*)(base + 2 * stride);
  float4 dw1 = *(const float4*)(base + 2 * stride + 4);
  float4 ww0 = *(const float4*)(base + 3 * stride);
  float4 ww1 = *(const float4*)(base + 3 * stride + 4);

  const int o = lane * 8;
  float pD0 = dot8(dd0, dd1, Dw + o)        + dot8(wd0, wd1, Dw + 256 + o);
  float pD1 = dot8(dd0, dd1, Dw + 512 + o)  + dot8(wd0, wd1, Dw + 768 + o);
  float pW0 = dot8(dw0, dw1, Ww + o)        + dot8(ww0, ww1, Ww + 256 + o);
  float pW1 = dot8(dw0, dw1, Ww + 512 + o)  + dot8(ww0, ww1, Ww + 768 + o);
#pragma unroll
  for (int m = 1; m < 32; m <<= 1) {
    pD0 += __shfl_xor(pD0, m, 32);
    pD1 += __shfl_xor(pD1, m, 32);
    pW0 += __shfl_xor(pW0, m, 32);
    pW1 += __shfl_xor(pW1, m, 32);
  }
  float lD0 = pD0 + Db[0], lD1 = pD1 + Db[1];
  float lW0 = pW0 + Wb[0], lW1 = pW1 + Wb[1];
  float mD = fmaxf(lD0, lD1), mW = fmaxf(lW0, lW1);
  float eD0 = expf(lD0 - mD), eD1 = expf(lD1 - mD);
  float eW0 = expf(lW0 - mW), eW1 = expf(lW1 - mW);
  float aD0 = eD0 / (eD0 + eD1), aD1 = eD1 / (eD0 + eD1);
  float aW0 = eW0 / (eW0 + eW1), aW1 = eW1 / (eW0 + eW1);

  bool tsel = mask[n] != 0;
  float4 o0, o1;
  if (tsel) {
    o0.x = dw0.x * aW0 + ww0.x * aW1; o0.y = dw0.y * aW0 + ww0.y * aW1;
    o0.z = dw0.z * aW0 + ww0.z * aW1; o0.w = dw0.w * aW0 + ww0.w * aW1;
    o1.x = dw1.x * aW0 + ww1.x * aW1; o1.y = dw1.y * aW0 + ww1.y * aW1;
    o1.z = dw1.z * aW0 + ww1.z * aW1; o1.w = dw1.w * aW0 + ww1.w * aW1;
  } else {
    o0.x = dd0.x * aD0 + wd0.x * aD1; o0.y = dd0.y * aD0 + wd0.y * aD1;
    o0.z = dd0.z * aD0 + wd0.z * aD1; o0.w = dd0.w * aD0 + wd0.w * aD1;
    o1.x = dd1.x * aD0 + wd1.x * aD1; o1.y = dd1.y * aD0 + wd1.y * aD1;
    o1.z = dd1.z * aD0 + wd1.z * aD1; o1.w = dd1.w * aD0 + wd1.w * aD1;
  }
  float* op = out + (size_t)n * HO + lane * 8;
  *(float4*)(op)     = o0;
  *(float4*)(op + 4) = o1;
}

extern "C" void kernel_launch(void* const* d_in, const int* in_sizes, int n_in,
                              void* d_out, int out_size, void* d_ws, size_t ws_size,
                              hipStream_t stream) {
  (void)in_sizes; (void)n_in; (void)out_size; (void)ws_size;
  const float* feature = (const float*)d_in[0];
  const int*   src     = (const int*)d_in[1];
  const int*   dst     = (const int*)d_in[2];
  const unsigned char* type_mask = (const unsigned char*)d_in[3];  // jax bool = 1B
  const float* Wl      = (const float*)d_in[4];
  const float* a_src   = (const float*)d_in[5];
  const float* a_dst   = (const float*)d_in[6];
  const float* bias_p  = (const float*)d_in[7];
  const float* Dw      = (const float*)d_in[8];
  const float* Db      = (const float*)d_in[9];
  const float* Ww      = (const float*)d_in[10];
  const float* Wb      = (const float*)d_in[11];
  float* out = (float*)d_out;

  char* ws = (char*)d_ws;
  size_t off = 0;
  auto carve = [&](size_t bytes) -> char* {
    char* p = ws + off;
    off += (bytes + 255) & ~(size_t)255;
    return p;
  };
  unsigned short* featB = (unsigned short*)carve((size_t)NN * IFT * 2);
  unsigned short* WB    = (unsigned short*)carve((size_t)NC * HO * IFT * 2);
  float* Wh    = (float*)carve((size_t)NC * NN * HO * 4);
  float* Hacc  = (float*)carve((size_t)NC * NN * HO * 4);
  float* el    = (float*)carve((size_t)NC * NN * NH * 4);
  float* er    = (float*)carve((size_t)NC * NN * NH * 4);
  unsigned* mb = (unsigned*)carve((size_t)NC * NN * NH * 4);
  float* den   = (float*)carve((size_t)NC * NN * NH * 4);
  float* eBuf  = (float*)carve((size_t)NC * NE * NH * 4);

  // f32 -> bf16 copies for WMMA operands
  k_cvt_bf16<<<(NN * IFT / 4 + 255) / 256, 256, 0, stream>>>(feature, featB, NN * IFT / 4);
  k_cvt_bf16<<<(NC * HO * IFT / 4 + 255) / 256, 256, 0, stream>>>(Wl, WB, NC * HO * IFT / 4);
  // init accumulators / segment buffers
  k_fill_u32<<<(NC * NN * HO + 255) / 256, 256, 0, stream>>>((unsigned*)Hacc, 0u, NC * NN * HO);
  k_fill_u32<<<(NC * NN * NH + 255) / 256, 256, 0, stream>>>(mb, 0x007FFFFFu /* enc(-inf) */, NC * NN * NH);
  k_fill_u32<<<(NC * NN * NH + 255) / 256, 256, 0, stream>>>((unsigned*)den, 0u, NC * NN * NH);
  // dense projection (WMMA bf16 -> f32, async double-buffered staging)
  k_gemm<<<dim3((NN + 63) / 64, HO / 64, NC), 256, 0, stream>>>(featB, WB, Wh);
  // attention scalars
  k_attn<<<dim3((NN + 7) / 8, NC), 256, 0, stream>>>(Wh, a_src, a_dst, el, er);
  // segment softmax over dst neighborhoods
  k_edge_max<<<dim3((NE + 255) / 256, NC), 256, 0, stream>>>(src, dst, el, er, eBuf, mb);
  k_decode_m<<<(NC * NN * NH + 255) / 256, 256, 0, stream>>>(mb, NC * NN * NH);
  k_edge_exp<<<dim3((NE + 255) / 256, NC), 256, 0, stream>>>(dst, eBuf, (const float*)mb, den);
  // weighted scatter-sum (channel-major so per-channel Wh stays L2-resident)
  k_edge_scatter<<<dim3((NE + 7) / 8, NC), 256, 0, stream>>>(src, dst, eBuf, den, Wh, Hacc);
  // residual + bias + ELU
  k_elu_bias<<<(NC * NN * HO / 4 + 255) / 256, 256, 0, stream>>>(Hacc, Wh, bias_p, NC * NN * HO / 4);
  // semantic-level aggregation + node-type select
  k_aggregate<<<(NN + 7) / 8, 256, 0, stream>>>(Hacc, Dw, Db, Ww, Wb, type_mask, out);
}
